// SelfAttention_4733053960408
// MI455X (gfx1250) — matrix operand order repaired and checked by reading, not compiled, not run on hardware
//
#include <hip/hip_runtime.h>
#include <hip/hip_bf16.h>

// ---------------------------------------------------------------------------
// Fused multi-head self-attention for MI455X (gfx1250), wave32 + WMMA bf16.
//   S=4096, D_MODEL=512, H=8, D_K=64
// Pipeline:
//   1) proj_qkv   : Q/K/V = x @ W{Q,K,V}  (bf16 out, Q pre-scaled by 1/8)
//   2) attn       : flash-attention per (64-query block, head), fp32 out
//                   K tile staged with GLOBAL_LOAD_ASYNC_TO_LDS_B128 (ASYNCcnt)
//   3) out_proj   : concat(out) @ WO -> d_out (fp32)
// ---------------------------------------------------------------------------

#define S_LEN   4096
#define DMODEL  512
#define NHEADS  8
#define DK      64

typedef __attribute__((ext_vector_type(16))) __bf16 v16bf;
typedef __attribute__((ext_vector_type(8)))  __bf16 bf16x8;
typedef __attribute__((ext_vector_type(8)))  float  v8f;

__device__ __forceinline__ v8f wmma_bf16(v16bf a, v16bf b, v8f c) {
  // (neg_a, A, neg_b, B, c_mod, C, reuse_a, reuse_b)
  return __builtin_amdgcn_wmma_f32_16x16x32_bf16(false, a, false, b,
                                                 (short)0, c, false, false);
}

// Async global->LDS copy of 16 bytes per lane, tracked by ASYNCcnt.
// LDS flat pointers carry the LDS byte offset in the low 32 bits.
__device__ __forceinline__ void async_g2l_b128(void* lds_ptr, const void* gptr) {
  const unsigned int lds_off = (unsigned int)(uintptr_t)lds_ptr;
  asm volatile("global_load_async_to_lds_b128 %0, %1, off"
               :: "v"(lds_off), "v"((unsigned long long)(uintptr_t)gptr)
               : "memory");
}
__device__ __forceinline__ void wait_async0() {
  asm volatile("s_wait_asynccnt 0x0" ::: "memory");
}

// -------------------------------------------------------------------------
// Kernel 1: QKV projection. One wave per (16-row tile, head, which).
// Computes a 16x64 bf16 output tile: 16 K-steps of 4 WMMAs.
// -------------------------------------------------------------------------
__global__ __launch_bounds__(32)
void proj_qkv_kernel(const float* __restrict__ x,
                     const float* __restrict__ WQ,
                     const float* __restrict__ WK,
                     const float* __restrict__ WV,
                     __bf16* __restrict__ Qb,
                     __bf16* __restrict__ Kb,
                     __bf16* __restrict__ Vb)
{
  const int lane = threadIdx.x & 31;
  const int l15  = lane & 15;
  const int lhi  = lane >> 4;          // 0 or 1 (half-wave)
  const int m0   = blockIdx.x * 16;
  const int h    = blockIdx.y;
  const int which = blockIdx.z;        // 0=Q 1=K 2=V

  const float* W = (which == 0) ? WQ : (which == 1) ? WK : WV;
  __bf16* Out    = (which == 0) ? Qb : (which == 1) ? Kb : Vb;
  const float scale = (which == 0) ? 0.125f : 1.0f;   // 1/sqrt(64) folded into Q
  W   += h * (DMODEL * DK);
  Out += h * (S_LEN * DK);

  v8f acc[4] = {};
  const int arow = m0 + l15;

  for (int k0 = 0; k0 < DMODEL; k0 += 32) {
    // A tile 16x32: lanes 0-15 hold K {0..15}; lanes 16-31 hold {16..31}
    v16bf a;
    {
      const float* ap = x + arow * DMODEL + k0 + lhi * 16;
      #pragma unroll
      for (int j = 0; j < 8; ++j) a[j]     = (__bf16)ap[j];
      #pragma unroll
      for (int j = 0; j < 8; ++j) a[8 + j] = (__bf16)ap[8 + j];
    }
    #pragma unroll
    for (int nt = 0; nt < 4; ++nt) {
      // B tile 32x16: lane = column, half-wave selects K 0..15 / 16..31
      const int col = nt * 16 + l15;
      const float* bp = W + (k0 + lhi * 16) * DK + col;
      v16bf b;
      #pragma unroll
      for (int j = 0; j < 16; ++j) b[j] = (__bf16)bp[j * DK];
      acc[nt] = wmma_bf16(a, b, acc[nt]);
    }
  }

  // C layout: VGPR r -> row r (lanes 0-15) / row r+8 (lanes 16-31), col = lane%16
  #pragma unroll
  for (int nt = 0; nt < 4; ++nt) {
    const int col = nt * 16 + l15;
    #pragma unroll
    for (int r = 0; r < 8; ++r) {
      const int row = m0 + r + lhi * 8;
      Out[row * DK + col] = (__bf16)(acc[nt][r] * scale);
    }
  }
}

// -------------------------------------------------------------------------
// Kernel 2: flash attention. Block = 128 threads (4 waves) per (64 q-rows, head).
// Each wave owns 16 query rows; iterates 64-key blocks staged in LDS.
// -------------------------------------------------------------------------
#define BQ 64
#define BK 64
#define KSTR 72   // LDS row stride (elements); 144B rows keep 16B alignment

__global__ __launch_bounds__(128)
void attn_kernel(const __bf16* __restrict__ Qb,
                 const __bf16* __restrict__ Kb,
                 const __bf16* __restrict__ Vb,
                 float* __restrict__ Ob)
{
  alignas(16) __shared__ __bf16 Kt[BK][KSTR];   // [key][dk]   row-major
  alignas(16) __shared__ __bf16 Vt[DK][KSTR];   // [dk][key]   transposed
  alignas(16) __shared__ __bf16 Pt[4][16][64];  // per-wave P tile (C->A relayout)

  const int tid  = threadIdx.x;
  const int lane = tid & 31;
  const int w    = tid >> 5;
  const int l15  = lane & 15;
  const int lhi  = lane >> 4;
  const int h    = blockIdx.y;
  const int q0   = blockIdx.x * BQ;

  const __bf16* Qh = Qb + h * (S_LEN * DK);
  const __bf16* Kh = Kb + h * (S_LEN * DK);
  const __bf16* Vh = Vb + h * (S_LEN * DK);

  // Load this wave's Q tile (16x64) into A-register layout, two K-chunks of 32.
  const int qrow = q0 + w * 16 + l15;
  v16bf qa[2];
  #pragma unroll
  for (int c = 0; c < 2; ++c) {
    const __bf16* qp = Qh + qrow * DK + c * 32 + lhi * 8;
    bf16x8 lo = *(const bf16x8*)(qp);
    bf16x8 hi = *(const bf16x8*)(qp + 16);
    #pragma unroll
    for (int j = 0; j < 8; ++j) { qa[c][j] = lo[j]; qa[c][8 + j] = hi[j]; }
  }

  float mrun[8], lrun[8];
  v8f o[4] = {};
  #pragma unroll
  for (int r = 0; r < 8; ++r) { mrun[r] = -__builtin_inff(); lrun[r] = 0.0f; }

  // Per-thread staging addresses: each thread owns half a row (32 elements).
  const int srow = tid >> 1;
  const int scb  = (tid & 1) * 32;

  for (int j0 = 0; j0 < S_LEN; j0 += BK) {
    // ---- cooperative staging ----
    {
      const __bf16* kp = Kh + (j0 + srow) * DK + scb;
      const __bf16* vp = Vh + (j0 + srow) * DK + scb;
      // K: async global->LDS, row-major, 4 x 16B per thread (ASYNCcnt path)
      #pragma unroll
      for (int j = 0; j < 4; ++j)
        async_g2l_b128(&Kt[srow][scb + j * 8], kp + j * 8);
      // V: through registers with transpose into Vt[dk][key]
      #pragma unroll
      for (int j = 0; j < 32; ++j)
        Vt[scb + j][srow] = vp[j];
      // prefetch next key-block into cache while this block computes
      if (j0 + BK < S_LEN) {
        __builtin_prefetch(kp + BK * DK, 0, 0);
        __builtin_prefetch(vp + BK * DK, 0, 0);
      }
    }
    wait_async0();       // drain this wave's async copies into LDS
    __syncthreads();     // publish LDS tiles to all waves

    // ---- S = Q K^T (16 x 64), 8 WMMAs ----
    v8f sacc[4] = {};
    #pragma unroll
    for (int c = 0; c < 2; ++c) {
      #pragma unroll
      for (int nt = 0; nt < 4; ++nt) {
        const int n = nt * 16 + l15;                 // key row = score column
        const __bf16* bp = &Kt[n][c * 32 + lhi * 8];
        bf16x8 lo = *(const bf16x8*)(bp);
        bf16x8 hi = *(const bf16x8*)(bp + 16);
        v16bf b;
        #pragma unroll
        for (int j = 0; j < 8; ++j) { b[j] = lo[j]; b[8 + j] = hi[j]; }
        sacc[nt] = wmma_bf16(qa[c], b, sacc[nt]);
      }
    }

    // ---- online softmax (rows live per VGPR slot, half-wave split) ----
    float alpha[8];
    #pragma unroll
    for (int r = 0; r < 8; ++r) {
      float vmax = fmaxf(fmaxf(sacc[0][r], sacc[1][r]),
                         fmaxf(sacc[2][r], sacc[3][r]));
      #pragma unroll
      for (int off = 1; off < 16; off <<= 1)
        vmax = fmaxf(vmax, __shfl_xor(vmax, off, 32));
      const float mnew = fmaxf(mrun[r], vmax);
      alpha[r] = __expf(mrun[r] - mnew);
      mrun[r]  = mnew;
      float rs = 0.0f;
      #pragma unroll
      for (int nt = 0; nt < 4; ++nt) {
        const float p = __expf(sacc[nt][r] - mnew);
        sacc[nt][r] = p;
        rs += p;
      }
      #pragma unroll
      for (int off = 1; off < 16; off <<= 1)
        rs += __shfl_xor(rs, off, 32);
      lrun[r] = lrun[r] * alpha[r] + rs;
    }
    #pragma unroll
    for (int nt = 0; nt < 4; ++nt)
      #pragma unroll
      for (int r = 0; r < 8; ++r)
        o[nt][r] *= alpha[r];

    // ---- P: C-layout regs -> bf16 in LDS (A-layout readback) ----
    #pragma unroll
    for (int nt = 0; nt < 4; ++nt)
      #pragma unroll
      for (int r = 0; r < 8; ++r)
        Pt[w][r + lhi * 8][nt * 16 + l15] = (__bf16)sacc[nt][r];
    __syncthreads();

    // ---- O += P @ V (16 x 64), 8 WMMAs ----
    #pragma unroll
    for (int c = 0; c < 2; ++c) {
      const __bf16* pp = &Pt[w][l15][c * 32 + lhi * 8];
      bf16x8 plo = *(const bf16x8*)(pp);
      bf16x8 phi = *(const bf16x8*)(pp + 16);
      v16bf pa;
      #pragma unroll
      for (int j = 0; j < 8; ++j) { pa[j] = plo[j]; pa[8 + j] = phi[j]; }
      #pragma unroll
      for (int nt = 0; nt < 4; ++nt) {
        const int ndk = nt * 16 + l15;               // output feature column
        const __bf16* vb = &Vt[ndk][c * 32 + lhi * 8];
        bf16x8 lo = *(const bf16x8*)(vb);
        bf16x8 hi = *(const bf16x8*)(vb + 16);
        v16bf b;
        #pragma unroll
        for (int j = 0; j < 8; ++j) { b[j] = lo[j]; b[8 + j] = hi[j]; }
        o[nt] = wmma_bf16(pa, b, o[nt]);
      }
    }
    __syncthreads();
  }

  // ---- epilogue: normalize, write fp32 concat layout [S][DMODEL] ----
  float invl[8];
  #pragma unroll
  for (int r = 0; r < 8; ++r) invl[r] = 1.0f / lrun[r];
  #pragma unroll
  for (int nt = 0; nt < 4; ++nt) {
    const int col = h * DK + nt * 16 + l15;
    #pragma unroll
    for (int r = 0; r < 8; ++r) {
      const int row = q0 + w * 16 + r + lhi * 8;
      Ob[row * DMODEL + col] = o[nt][r] * invl[r];
    }
  }
}

// -------------------------------------------------------------------------
// Kernel 3: output projection  out = Ob @ WO  (4096x512x512), fp32 out.
// One wave per (16-row tile, 64-col block).
// -------------------------------------------------------------------------
__global__ __launch_bounds__(32)
void out_proj_kernel(const float* __restrict__ A,    // [S][DMODEL] fp32
                     const float* __restrict__ WO,   // [DMODEL][DMODEL] fp32
                     float* __restrict__ Out)        // [S][DMODEL] fp32
{
  const int lane = threadIdx.x & 31;
  const int l15  = lane & 15;
  const int lhi  = lane >> 4;
  const int m0   = blockIdx.x * 16;
  const int n0   = blockIdx.y * 64;

  v8f acc[4] = {};
  const int arow = m0 + l15;

  for (int k0 = 0; k0 < DMODEL; k0 += 32) {
    v16bf a;
    {
      const float* ap = A + arow * DMODEL + k0 + lhi * 16;
      #pragma unroll
      for (int j = 0; j < 8; ++j) a[j]     = (__bf16)ap[j];
      #pragma unroll
      for (int j = 0; j < 8; ++j) a[8 + j] = (__bf16)ap[8 + j];
    }
    #pragma unroll
    for (int nt = 0; nt < 4; ++nt) {
      const int col = n0 + nt * 16 + l15;
      const float* bp = WO + (k0 + lhi * 16) * DMODEL + col;
      v16bf b;
      #pragma unroll
      for (int j = 0; j < 16; ++j) b[j] = (__bf16)bp[j * DMODEL];
      acc[nt] = wmma_bf16(a, b, acc[nt]);
    }
  }

  #pragma unroll
  for (int nt = 0; nt < 4; ++nt) {
    const int col = n0 + nt * 16 + l15;
    #pragma unroll
    for (int r = 0; r < 8; ++r) {
      const int row = m0 + r + lhi * 8;
      Out[row * DMODEL + col] = acc[nt][r];
    }
  }
}

// -------------------------------------------------------------------------
extern "C" void kernel_launch(void* const* d_in, const int* in_sizes, int n_in,
                              void* d_out, int out_size, void* d_ws, size_t ws_size,
                              hipStream_t stream)
{
  const float* x  = (const float*)d_in[0];
  const float* WQ = (const float*)d_in[1];
  const float* WK = (const float*)d_in[2];
  const float* WV = (const float*)d_in[3];
  const float* WO = (const float*)d_in[4];

  // Workspace: Qb|Kb|Vb (bf16, 4 MiB each) + Ob (fp32, 8 MiB) = 20 MiB
  const size_t qkv_elems = (size_t)NHEADS * S_LEN * DK;
  __bf16* Qb = (__bf16*)d_ws;
  __bf16* Kb = Qb + qkv_elems;
  __bf16* Vb = Kb + qkv_elems;
  float*  Ob = (float*)(Vb + qkv_elems);

  {
    dim3 grid(S_LEN / 16, NHEADS, 3);
    proj_qkv_kernel<<<grid, 32, 0, stream>>>(x, WQ, WK, WV, Qb, Kb, Vb);
  }
  {
    dim3 grid(S_LEN / BQ, NHEADS);
    attn_kernel<<<grid, 128, 0, stream>>>(Qb, Kb, Vb, Ob);
  }
  {
    dim3 grid(S_LEN / 16, DMODEL / 64);
    out_proj_kernel<<<grid, 32, 0, stream>>>(Ob, WO, (float*)d_out);
  }
}
